// LocalTransformerAggregation_29841432773196
// MI455X (gfx1250) — matrix operand order repaired and checked by reading, not compiled, not run on hardware
//
#include <hip/hip_runtime.h>
#include <hip/hip_bf16.h>

typedef _Float16 half_t;
typedef __attribute__((ext_vector_type(16))) _Float16 v16h;
typedef __attribute__((ext_vector_type(8)))  _Float16 v8h;
typedef __attribute__((ext_vector_type(8)))  float    v8f;
typedef __attribute__((ext_vector_type(4)))  unsigned int v4u;
typedef __attribute__((ext_vector_type(4)))  int v4i;
typedef __attribute__((ext_vector_type(8)))  int v8i;

#define IN_C      128
#define OUT_C     256
#define HEADS     8
#define HEAD_DIM  32
#define KNEIGH    16
#define TILE_PTS  16

// ---------------- workspace layout (bytes) ----------------
// Per-head weight bundles: [Wp2 16 blk | Wk 8 blk | Wv 8 blk] = 32 blocks * 1KB = 32KB/head.
static constexpr size_t OFF_BUND = 0;        // 8 * 32768 = 256KB
static constexpr size_t OFF_WQ   = 262144;   // 64KB (std layout, read-once)
static constexpr size_t OFF_WR   = 327680;   // 64KB
static constexpr size_t OFF_WO   = 393216;   // 128KB
static constexpr size_t OFF_WP1  = 524288;   // 16KB (TDM-staged)
static constexpr size_t OFF_BQ   = 540672;   // f32[256] each
static constexpr size_t OFF_BP1  = 541696;
static constexpr size_t OFF_BKP  = 542720;   // bk + bp2
static constexpr size_t OFF_BVP  = 543744;   // bv + bp2
static constexpr size_t OFF_BOR  = 544768;   // bo + br
static constexpr size_t WS_NEEDED = 545792;

// ---------------- LDS layout (bytes) ----------------
static constexpr int S_NEIGH   = 0;        // 256x128 f16 = 65536
static constexpr int S_HIDDEN  = 65536;    // 256x256 f16 = 131072
static constexpr int S_POSIN   = 196608;   // 256x32  f16 = 16384
static constexpr int S_Q       = 212992;   // 16x256  f32 = 16384
static constexpr int S_KH      = 229376;   // 256x32  f16 = 16384
static constexpr int S_VH      = 245760;   // 256x32  f16 = 16384
static constexpr int S_CF      = 262144;   // 16x128  f16 = 4096
static constexpr int S_AO      = 266240;   // 16x256  f16 = 8192
static constexpr int S_STAGE   = 274432;   // 32KB TDM weight stage
static constexpr int SMEM_BYTES = 307200;  // <= 320KB WGP limit

// ---------------- WMMA helpers ----------------
static __device__ inline v8f wmma_f16(v16h a, v16h b, v8f c) {
  return __builtin_amdgcn_wmma_f32_16x16x32_f16(
      false, a, false, b, (short)0, c, false, false);
}

// A operand: 16x32 f16 tile from row-major LDS (stride in halves).
static __device__ inline v16h load_A(const half_t* base, int stride, int lane) {
  const int m  = lane & 15;
  const int hi = (lane >> 4) * 16;
  const half_t* r = base + m * stride + hi;
  v8h lo = *(const v8h*)(r);
  v8h hh = *(const v8h*)(r + 8);
  v16h o;
#pragma unroll
  for (int i = 0; i < 8; ++i) { o[i] = lo[i]; o[8 + i] = hh[i]; }
  return o;
}

// B operand: 32x16 f16 tile pre-packed so each lane reads 16 contiguous halves (32B).
static __device__ inline v16h load_B(const half_t* blk, int lane) {
  return *(const v16h*)(blk + lane * 16);
}

// ---------------- TDM staging: contiguous copy of n8 8-byte units ----------------
static __device__ inline void tdm_stage(const char* gsrc, char* lds_dst, unsigned n8) {
#if __has_builtin(__builtin_amdgcn_tensor_load_to_lds)
  const unsigned long long ga = (unsigned long long)(uintptr_t)gsrc;
  v4u g0;
  g0[0] = 1u;                                   // count=1, user descriptor
  g0[1] = (unsigned)(uintptr_t)lds_dst;         // lds_addr (flat low 32 = LDS offset)
  g0[2] = (unsigned)(ga & 0xFFFFFFFFu);         // global_addr[31:0]
  g0[3] = (unsigned)((ga >> 32) & 0x1FFFFFFu) | (2u << 30);  // addr[56:32] | type=2
  v8i g1;
  g1[0] = (int)(3u << 16);                      // wg_mask=0, data_size=8B
  g1[1] = (int)((n8 & 0xFFFFu) << 16);          // tensor_dim0[15:0] in [31:16]
  g1[2] = (int)(((n8 >> 16) & 0xFFFFu) | (1u << 16)); // dim0 hi | tensor_dim1=1
  g1[3] = (int)((n8 & 0xFFFFu) << 16);          // tensor_dim1 hi=0 | tile_dim0=n8
  g1[4] = 1;                                    // tile_dim1=1, tile_dim2=0
  g1[5] = (int)n8;                              // tensor_dim0_stride lo
  g1[6] = 0;                                    // stride0 hi | stride1 lo
  g1[7] = 0;                                    // stride1 hi
  v4i z4 = {0, 0, 0, 0};
#if defined(__clang_major__) && (__clang_major__ >= 23)
  v8i z8 = {0, 0, 0, 0, 0, 0, 0, 0};
  __builtin_amdgcn_tensor_load_to_lds(g0, g1, z4, z4, z8, 0);
#else
  __builtin_amdgcn_tensor_load_to_lds(g0, g1, z4, z4, 0);
#endif
#else
  const unsigned lane = threadIdx.x & 31;       // fallback: wave copy
  for (unsigned i = lane; i < n8; i += 32)
    ((unsigned long long*)lds_dst)[i] = ((const unsigned long long*)gsrc)[i];
#endif
}

static __device__ inline void tdm_wait() {
#if __has_builtin(__builtin_amdgcn_tensor_load_to_lds)
  __builtin_amdgcn_s_wait_tensorcnt(0);
#endif
}

// ---------------- packing kernels ----------------
// Standard layout: block (kt,nt) at bi=kt*16+nt; lane l: n=nt*16+(l&15), k=kt*32+(l>>4)*16+j.
__global__ void pack_weight(const float* __restrict__ W, half_t* __restrict__ dst, int Kdim) {
  const int bi = blockIdx.x;
  const int kt = bi >> 4, nt = bi & 15;
  const int l  = threadIdx.x;
  half_t* d = dst + (size_t)bi * 512 + l * 16;
#pragma unroll
  for (int j = 0; j < 16; ++j) {
    const int k = kt * 32 + (l >> 4) * 16 + j;
    const int n = nt * 16 + (l & 15);
    d[j] = (half_t)((k < Kdim) ? W[(size_t)k * OUT_C + n] : 0.f);
  }
}

// Per-head bundle layout: head h = nt>>1, ct = nt&1; block slot = partOff + kt*2 + ct.
__global__ void pack_weight_heads(const float* __restrict__ W, half_t* __restrict__ bund,
                                  int Kdim, int partOff) {
  const int bi = blockIdx.x;
  const int kt = bi >> 4, nt = bi & 15;
  const int l  = threadIdx.x;
  const int h  = nt >> 1, ct = nt & 1;
  half_t* d = bund + (size_t)h * 16384 + (size_t)(partOff + kt * 2 + ct) * 512 + l * 16;
#pragma unroll
  for (int j = 0; j < 16; ++j) {
    const int k = kt * 32 + (l >> 4) * 16 + j;
    const int n = nt * 16 + (l & 15);
    d[j] = (half_t)((k < Kdim) ? W[(size_t)k * OUT_C + n] : 0.f);
  }
}

__global__ void pack_bias(const float* __restrict__ a, const float* __restrict__ b,
                          float* __restrict__ d) {
  const int t = threadIdx.x;
  d[t] = a[t] + (b ? b[t] : 0.f);
}

// ---------------- fused main kernel ----------------
__global__ __launch_bounds__(256) void lta_main(
    const float* __restrict__ cxyz,  const float* __restrict__ cfeat,
    const float* __restrict__ nxyz,  const float* __restrict__ nfeat,
    const char*  __restrict__ ws,    float* __restrict__ out) {
  extern __shared__ __align__(16) char smem[];
  half_t* sNeigh = (half_t*)(smem + S_NEIGH);
  half_t* sHid   = (half_t*)(smem + S_HIDDEN);
  half_t* sPos   = (half_t*)(smem + S_POSIN);
  float*  sQ     = (float*) (smem + S_Q);
  half_t* sKh    = (half_t*)(smem + S_KH);
  half_t* sVh    = (half_t*)(smem + S_VH);
  half_t* sCF    = (half_t*)(smem + S_CF);
  half_t* sAO    = (half_t*)(smem + S_AO);
  half_t* sStg   = (half_t*)(smem + S_STAGE);

  const half_t* WqP = (const half_t*)(ws + OFF_WQ);
  const half_t* WrP = (const half_t*)(ws + OFF_WR);
  const half_t* WoP = (const half_t*)(ws + OFF_WO);
  const float*  bq  = (const float*)(ws + OFF_BQ);
  const float*  bp1 = (const float*)(ws + OFF_BP1);
  const float*  bkp = (const float*)(ws + OFF_BKP);
  const float*  bvp = (const float*)(ws + OFF_BVP);
  const float*  bor = (const float*)(ws + OFF_BOR);

  const int tid  = threadIdx.x;
  const int lane = tid & 31;
  const int w    = tid >> 5;          // wave id 0..7 (wave32)
  const int pbase = blockIdx.x * TILE_PTS;

  // Stage Wp1 via TDM (overlaps with phase-0 input staging below).
  if (w == 0) tdm_stage(ws + OFF_WP1, (char*)sStg, 16384 / 8);
  if (tid < 64) __builtin_prefetch(ws + OFF_WO + (size_t)tid * 2048, 0, 1);

  // ---- Phase 0: stage inputs to LDS as f16 ----
  {
    const float* src = cfeat + (size_t)pbase * IN_C;     // 16x128
#pragma unroll
    for (int i = 0; i < 2; ++i) {
      const int f4 = tid * 2 + i;
      const float4 v = ((const float4*)src)[f4];
      half_t* d = sCF + f4 * 4;
      d[0] = (half_t)v.x; d[1] = (half_t)v.y; d[2] = (half_t)v.z; d[3] = (half_t)v.w;
    }
    const float* nsrc = nfeat + (size_t)pbase * KNEIGH * IN_C;  // 256x128 contiguous
#pragma unroll
    for (int i = 0; i < 32; ++i) {
      const int f4 = tid + 256 * i;
      const float4 v = ((const float4*)nsrc)[f4];
      half_t* d = sNeigh + f4 * 4;
      d[0] = (half_t)v.x; d[1] = (half_t)v.y; d[2] = (half_t)v.z; d[3] = (half_t)v.w;
    }
    const int r = tid;                                   // neighbor row 0..255
    const int p = r >> 4;
    const float* nx = nxyz + ((size_t)pbase * KNEIGH + r) * 3;
    const float* cx = cxyz + (size_t)(pbase + p) * 3;
    const float dx = nx[0] - cx[0], dy = nx[1] - cx[1], dz = nx[2] - cx[2];
    const float dist = sqrtf(dx * dx + dy * dy + dz * dz);
    half_t* d = sPos + r * 32;
    d[0] = (half_t)dx; d[1] = (half_t)dy; d[2] = (half_t)dz; d[3] = (half_t)dist;
#pragma unroll
    for (int j = 4; j < 32; ++j) d[j] = (half_t)0.f;
  }
  if (w == 0) tdm_wait();
  __syncthreads();

  // ---- Phase 1a: q = cf @ Wq + bq  (16x256 f32 in LDS) ----
#pragma unroll
  for (int i = 0; i < 2; ++i) {
    const int nt = w * 2 + i;
    v8f acc = {};
#pragma unroll
    for (int c = 0; c < 4; ++c)
      acc = wmma_f16(load_A(sCF + c * 32, IN_C, lane),
                     load_B(WqP + (size_t)(c * 16 + nt) * 512, lane), acc);
    const int n = lane & 15, mh = (lane >> 4) * 8;
    const float b = bq[nt * 16 + n];
#pragma unroll
    for (int r = 0; r < 8; ++r) sQ[(mh + r) * OUT_C + nt * 16 + n] = acc[r] + b;
  }

  // ---- Phase 1b: hidden = relu(pos_in @ Wp1 + bp1), Wp1 from TDM stage ----
#pragma unroll
  for (int i = 0; i < 32; ++i) {
    const int tile = w * 32 + i;
    const int rt = tile >> 4, nt = tile & 15;
    v8f acc = {};
    acc = wmma_f16(load_A(sPos + rt * 16 * 32, 32, lane),
                   load_B(sStg + nt * 512, lane), acc);
    const int n = lane & 15, mh = (lane >> 4) * 8;
    const float b = bp1[nt * 16 + n];
#pragma unroll
    for (int r = 0; r < 8; ++r) {
      float x = acc[r] + b;
      x = x > 0.f ? x : 0.f;
      sHid[(rt * 16 + mh + r) * OUT_C + nt * 16 + n] = (half_t)x;
    }
  }
  __syncthreads();

  // ---- Phase 2: per head, TDM-stage its 32KB weight bundle, then k/v WMMA + attention ----
  for (int h = 0; h < HEADS; ++h) {
    if (w == 0) {
      tdm_stage(ws + OFF_BUND + (size_t)h * 32768, (char*)sStg, 32768 / 8);
      tdm_wait();
    }
    __syncthreads();   // bundle ready; previous head's reads of sStg finished earlier

#pragma unroll
    for (int i = 0; i < 4; ++i) {
      const int a_ = i * 8 + w;          // 0..31
      const int rt = a_ >> 1, ct = a_ & 1;
      const int nt = 2 * h + ct;
      const half_t* hbase = sHid + rt * 16 * OUT_C;
      v8f pe = {};
#pragma unroll
      for (int c = 0; c < 8; ++c)        // pe = hidden @ Wp2 slice (staged)
        pe = wmma_f16(load_A(hbase + c * 32, OUT_C, lane),
                      load_B(sStg + (c * 2 + ct) * 512, lane), pe);
      const half_t* nbase = sNeigh + rt * 16 * IN_C;
      v8f ak = pe, av = pe;
#pragma unroll
      for (int c = 0; c < 4; ++c) {      // k/v chained off pe
        v16h A = load_A(nbase + c * 32, IN_C, lane);
        ak = wmma_f16(A, load_B(sStg + (16 + c * 2 + ct) * 512, lane), ak);
        av = wmma_f16(A, load_B(sStg + (24 + c * 2 + ct) * 512, lane), av);
      }
      const int n = lane & 15, mh = (lane >> 4) * 8;
      const float bk_ = bkp[nt * 16 + n], bv_ = bvp[nt * 16 + n];
#pragma unroll
      for (int r = 0; r < 8; ++r) {
        const int row = rt * 16 + mh + r;
        sKh[row * HEAD_DIM + ct * 16 + n] = (half_t)(ak[r] + bk_);
        sVh[row * HEAD_DIM + ct * 16 + n] = (half_t)(av[r] + bv_);
      }
    }
    __syncthreads();

    // softmax attention over K=16 neighbors (one point per 16-lane group)
    {
      const int p = tid >> 4, j = tid & 15;
      const float* qrow = sQ + p * OUT_C + h * HEAD_DIM;
      const half_t* krow = sKh + (p * 16 + j) * HEAD_DIM;
      float s = 0.f;
#pragma unroll
      for (int d = 0; d < HEAD_DIM; ++d) s += qrow[d] * (float)krow[d];
      s *= 0.17677669529663689f;          // 1/sqrt(32)
      float mx = s;
#pragma unroll
      for (int o = 8; o > 0; o >>= 1) mx = fmaxf(mx, __shfl_xor(mx, o, 16));
      const float e = __expf(s - mx);
      float sum = e;
#pragma unroll
      for (int o = 8; o > 0; o >>= 1) sum += __shfl_xor(sum, o, 16);
      const float attn = e / sum;

      float a0 = 0.f, a1 = 0.f;
#pragma unroll
      for (int jj = 0; jj < 16; ++jj) {
        const float aj = __shfl(attn, jj, 16);
        const half_t* vr = sVh + (p * 16 + jj) * HEAD_DIM;
        a0 += aj * (float)vr[j];
        a1 += aj * (float)vr[j + 16];
      }
      sAO[p * OUT_C + h * HEAD_DIM + j]      = (half_t)a0;
      sAO[p * OUT_C + h * HEAD_DIM + j + 16] = (half_t)a1;
    }
    __syncthreads();
  }

  // ---- Phase 3: out = attnOut @ Wo + cf @ Wr + (bo+br), leaky_relu(0.2) ----
#pragma unroll
  for (int i = 0; i < 2; ++i) {
    const int nt = w * 2 + i;
    v8f acc = {};
#pragma unroll
    for (int c = 0; c < 8; ++c)
      acc = wmma_f16(load_A(sAO + c * 32, OUT_C, lane),
                     load_B(WoP + (size_t)(c * 16 + nt) * 512, lane), acc);
#pragma unroll
    for (int c = 0; c < 4; ++c)
      acc = wmma_f16(load_A(sCF + c * 32, IN_C, lane),
                     load_B(WrP + (size_t)(c * 16 + nt) * 512, lane), acc);
    const int n = lane & 15, mh = (lane >> 4) * 8;
    const float b = bor[nt * 16 + n];
#pragma unroll
    for (int r = 0; r < 8; ++r) {
      float x = acc[r] + b;
      x = x >= 0.f ? x : 0.2f * x;
      out[(size_t)(pbase + mh + r) * OUT_C + nt * 16 + n] = x;
    }
  }
}

// ---------------- host ----------------
extern "C" void kernel_launch(void* const* d_in, const int* in_sizes, int n_in,
                              void* d_out, int out_size, void* d_ws, size_t ws_size,
                              hipStream_t stream) {
  (void)n_in; (void)out_size;
  if (ws_size < WS_NEEDED) return;

  const float* cxyz  = (const float*)d_in[0];
  const float* cfeat = (const float*)d_in[1];
  const float* nxyz  = (const float*)d_in[2];
  const float* nfeat = (const float*)d_in[3];
  const float* Wq  = (const float*)d_in[4];   const float* bq  = (const float*)d_in[5];
  const float* Wk  = (const float*)d_in[6];   const float* bk  = (const float*)d_in[7];
  const float* Wv  = (const float*)d_in[8];   const float* bv  = (const float*)d_in[9];
  const float* Wp1 = (const float*)d_in[10];  const float* bp1 = (const float*)d_in[11];
  const float* Wp2 = (const float*)d_in[12];  const float* bp2 = (const float*)d_in[13];
  const float* Wo  = (const float*)d_in[14];  const float* bo  = (const float*)d_in[15];
  const float* Wr  = (const float*)d_in[16];  const float* br  = (const float*)d_in[17];
  char* ws = (char*)d_ws;

  half_t* bund = (half_t*)(ws + OFF_BUND);
  pack_weight_heads<<<128, 32, 0, stream>>>(Wp2, bund, OUT_C, 0);   // Wp2 -> slots 0..15
  pack_weight_heads<<<64,  32, 0, stream>>>(Wk,  bund, IN_C, 16);   // Wk  -> slots 16..23
  pack_weight_heads<<<64,  32, 0, stream>>>(Wv,  bund, IN_C, 24);   // Wv  -> slots 24..31
  pack_weight<<<64,  32, 0, stream>>>(Wq,  (half_t*)(ws + OFF_WQ),  IN_C);
  pack_weight<<<64,  32, 0, stream>>>(Wr,  (half_t*)(ws + OFF_WR),  IN_C);
  pack_weight<<<128, 32, 0, stream>>>(Wo,  (half_t*)(ws + OFF_WO),  OUT_C);
  pack_weight<<<16,  32, 0, stream>>>(Wp1, (half_t*)(ws + OFF_WP1), 4);
  pack_bias<<<1, 256, 0, stream>>>(bq,  nullptr, (float*)(ws + OFF_BQ));
  pack_bias<<<1, 256, 0, stream>>>(bp1, nullptr, (float*)(ws + OFF_BP1));
  pack_bias<<<1, 256, 0, stream>>>(bk,  bp2,     (float*)(ws + OFF_BKP));
  pack_bias<<<1, 256, 0, stream>>>(bv,  bp2,     (float*)(ws + OFF_BVP));
  pack_bias<<<1, 256, 0, stream>>>(bo,  br,      (float*)(ws + OFF_BOR));

  const int BN   = in_sizes[1] / IN_C;       // 16384
  const int nblk = BN / TILE_PTS;            // 1024
  hipFuncSetAttribute((const void*)lta_main,
                      hipFuncAttributeMaxDynamicSharedMemorySize, SMEM_BYTES);
  lta_main<<<nblk, 256, SMEM_BYTES, stream>>>(cxyz, cfeat, nxyz, nfeat, ws, (float*)d_out);
}
